// StyleTransformerEncoderBlock_69217692942501
// MI455X (gfx1250) — matrix operand order repaired and checked by reading, not compiled, not run on hardware
//
#include <hip/hip_runtime.h>

typedef __bf16 bf16_t;
typedef __attribute__((ext_vector_type(16))) __bf16 v16bf;
typedef __attribute__((ext_vector_type(8)))  __bf16 v8bf;
typedef __attribute__((ext_vector_type(8)))  float  v8f;

#define DIMC   512
#define HEADS  8
#define HD     64
#define NWIN   64          /* tokens per window */
#define NWPB   64          /* windows per batch image */
#define BATCH  16
#define MTOT   (BATCH*NWPB*NWIN)   /* 65536 rows */
#define HID    2048
#define SHIFT  4
#define SCQ    0.125f      /* HD^-0.5 */
#define LPAD   72          /* padded bf16 LDS row stride (128B row + 16B pad) */
#define SPAD   65          /* padded f32 LDS row stride for scores */

__device__ __forceinline__ v16bf cat16(v8bf a, v8bf b) {
  return __builtin_shufflevector(a, b, 0,1,2,3,4,5,6,7,8,9,10,11,12,13,14,15);
}

// window row (b, wh, ww, pi, pj) -> image row after reverse-shift (+SHIFT)
__device__ __forceinline__ int winrev_row(int row) {
  int b   = row >> 12;
  int rem = row & 4095;
  int wh  = rem >> 9;
  int ww  = (rem >> 6) & 7;
  int pi  = (rem >> 3) & 7;
  int pj  = rem & 7;
  int h = (wh * 8 + pi + SHIFT) & 63;
  int w = (ww * 8 + pj + SHIFT) & 63;
  return (b << 12) + h * 64 + w;
}

// Swin shifted-window region id for a (shifted-image) coordinate
__device__ __forceinline__ int region(int p) { return (p < 56) ? 0 : ((p < 60) ? 1 : 2); }

// ---------------------------------------------------------------------------
// Tensor Data Mover: DMA a 64x64 bf16 tile (row stride 512 elems) into LDS,
// with hardware padding: every 32 DWORDs (one 128B row) insert 4 DWORDs (16B)
// -> LDS row stride 144B = 72 bf16 elements (bank-conflict-free fragments).
// ---------------------------------------------------------------------------
#if __has_builtin(__builtin_amdgcn_tensor_load_to_lds)
#define HAVE_TDM 1
typedef unsigned int u32x4 __attribute__((ext_vector_type(4)));
typedef int          i32x4 __attribute__((ext_vector_type(4)));
typedef int          i32x8 __attribute__((ext_vector_type(8)));

__device__ __forceinline__ void tdm_load_tile64x64(const bf16_t* gsrc, void* lds_dst) {
  unsigned long long ga = (unsigned long long)(uintptr_t)gsrc;
  unsigned lds = (unsigned)(uintptr_t)lds_dst;
  u32x4 g0;
  g0[0] = 1u;                                    // count=1 (valid user D#)
  g0[1] = lds;                                   // lds_addr
  g0[2] = (unsigned)ga;                          // global_addr[31:0]
  g0[3] = (unsigned)((ga >> 32) & 0x01ffffffu)   // global_addr[56:32]
        | 0x80000000u;                           // type=2 ("image")
  i32x8 g1;
  g1[0] = (int)((1u << 16)                       // data_size = 2 bytes
              | (1u << 20)                       // pad_enable
              | (4u << 22)                       // pad_interval: 32 DWORDs
              | (3u << 25));                     // pad_amount:   4 DWORDs
  g1[1] = (int)(64u << 16);                      // tensor_dim0 = 64
  g1[2] = (int)(64u << 16);                      // tensor_dim1 = 64
  g1[3] = (int)(64u << 16);                      // tile_dim0   = 64
  g1[4] = 64;                                    // tile_dim1=64, tile_dim2=0
  g1[5] = 512;                                   // tensor_dim0_stride = 512
  g1[6] = 0;
  g1[7] = 0;
  i32x4 z4 = {0, 0, 0, 0};
  i32x8 z8 = {0, 0, 0, 0, 0, 0, 0, 0};
  __builtin_amdgcn_tensor_load_to_lds(g0, g1, z4, z4, z8, 0);
}
#else
#define HAVE_TDM 0
#endif

// ---------------------------------------------------------------------------
// Weight convert + transpose: W(K x N) f32 -> WT(N x K) bf16
// ---------------------------------------------------------------------------
__global__ __launch_bounds__(256) void k_wt(const float* __restrict__ w,
                                            bf16_t* __restrict__ wt,
                                            int K, int N) {
  size_t i = (size_t)blockIdx.x * 256 + threadIdx.x;
  if (i >= (size_t)K * N) return;
  int n = (int)(i / K);
  int k = (int)(i % K);
  wt[i] = (bf16_t)w[(size_t)k * N + n];
}

// ---------------------------------------------------------------------------
// Cyclic shift (-4,-4) + window partition + f32->bf16
// ---------------------------------------------------------------------------
__global__ __launch_bounds__(256) void k_part(const float* __restrict__ in,
                                              bf16_t* __restrict__ out) {
  size_t i = (size_t)blockIdx.x * 256 + threadIdx.x;   // m*512 + c
  int c = (int)(i & 511);
  int m = (int)(i >> 9);
  int b   = m >> 12;
  int rem = m & 4095;
  int wh  = rem >> 9;
  int ww  = (rem >> 6) & 7;
  int pi  = (rem >> 3) & 7;
  int pj  = rem & 7;
  int h = (wh * 8 + pi + SHIFT) & 63;
  int w = (ww * 8 + pj + SHIFT) & 63;
  out[i] = (bf16_t)in[((size_t)b * 4096 + h * 64 + w) * DIMC + c];
}

// ---------------------------------------------------------------------------
// WMMA GEMM: C[M,Nout] = A[M,K](bf16) @ WT(Nout x K, bf16, pre-transposed) + bias
// 256 threads = 8 waves (2x4); block tile 64x256; wave tile 32x64.
// mode==1: rows stored through window-reverse permutation.
// ---------------------------------------------------------------------------
__global__ __launch_bounds__(256)
void k_gemm(const bf16_t* __restrict__ A, const bf16_t* __restrict__ WT,
            const float* __restrict__ bias, int K, int Nout,
            bf16_t* __restrict__ outB, float* __restrict__ outF,
            const float* __restrict__ resid,
            int mode, int relu, float oscale) {
  const int tid   = threadIdx.x;
  const int wave  = tid >> 5;
  const int lane  = tid & 31;
  const int wm    = wave >> 2;          // 0..1
  const int wn    = wave & 3;           // 0..3
  const int mbase = blockIdx.y * 64 + wm * 32;
  const int nbase = blockIdx.x * 256 + wn * 64;
  const int lhalf = (lane >= 16) ? 1 : 0;
  const int l15   = lane & 15;

  v8f acc[2][4] = {};

  for (int k0 = 0; k0 < K; k0 += 32) {
    v16bf afrag[2], bfrag[4];
#pragma unroll
    for (int t = 0; t < 2; ++t) {
      // A fragment: lanes 0-15 hold K{0-7,16-23}; lanes 16-31 hold K{8-15,24-31}
      const bf16_t* pa = A + (size_t)(mbase + t * 16 + l15) * K + k0 + lhalf * 8;
      afrag[t] = cat16(*(const v8bf*)pa, *(const v8bf*)(pa + 16));
      __builtin_prefetch(pa + 128, 0, 0);   // speculative; dropped if invalid
    }
#pragma unroll
    for (int u = 0; u < 4; ++u) {
      // B fragment: lanes 0-15 hold K{0-7,16-23} of column n; lanes 16-31 K{8-15,24-31}
      const bf16_t* pb = WT + (size_t)(nbase + u * 16 + l15) * K + k0 + lhalf * 8;
      bfrag[u] = cat16(*(const v8bf*)pb, *(const v8bf*)(pb + 16));
    }
#pragma unroll
    for (int t = 0; t < 2; ++t)
#pragma unroll
      for (int u = 0; u < 4; ++u)
        acc[t][u] = __builtin_amdgcn_wmma_f32_16x16x32_bf16(
            false, afrag[t], false, bfrag[u], (short)0, acc[t][u], false, false);
  }

#pragma unroll
  for (int t = 0; t < 2; ++t)
#pragma unroll
    for (int u = 0; u < 4; ++u)
#pragma unroll
      for (int r = 0; r < 8; ++r) {
        int row = mbase + t * 16 + lhalf * 8 + r;
        int col = nbase + u * 16 + l15;
        float f = (acc[t][u][r] + bias[col]) * oscale;
        if (relu) f = fmaxf(f, 0.0f);
        int orow = (mode == 1) ? winrev_row(row) : row;
        size_t idx = (size_t)orow * Nout + col;
        if (resid) f += resid[idx];
        if (outF)  outF[idx] = f;
        if (outB)  outB[idx] = (bf16_t)f;
      }
}

// ---------------------------------------------------------------------------
// Windowed attention: one block = one (window, head). 128 threads = 4 waves.
// TDM stages Q,K,V,Vs,Vh tiles into LDS (padded rows); the three V DMAs
// overlap with score compute + softmax.
// ---------------------------------------------------------------------------
__global__ __launch_bounds__(128)
void k_attn(const bf16_t* __restrict__ q, const bf16_t* __restrict__ kk,
            const bf16_t* __restrict__ v, const bf16_t* __restrict__ vs,
            const bf16_t* __restrict__ vh,
            const float* __restrict__ rpb,
            bf16_t* __restrict__ o, bf16_t* __restrict__ os,
            bf16_t* __restrict__ oh) {
  __shared__ __align__(16) float  S[NWIN * SPAD];      // scores (padded)
  __shared__ __align__(16) bf16_t P[NWIN * LPAD];      // probs bf16 (padded)
  __shared__ __align__(16) bf16_t Ql[NWIN * LPAD];
  __shared__ __align__(16) bf16_t Kl[NWIN * LPAD];
  __shared__ __align__(16) bf16_t Vrow[3][NWIN * LPAD];
  __shared__ __align__(16) bf16_t VT[NWIN * LPAD];

  const int blk  = blockIdx.x;          // 0..8191
  const int w    = blk >> 3;            // window 0..1023
  const int hh   = blk & 7;             // head
  const int tid  = threadIdx.x;
  const int wave = tid >> 5;
  const int lane = tid & 31;
  const int lhalf = (lane >= 16) ? 1 : 0;
  const int l15   = lane & 15;
  const size_t base = (size_t)w * NWIN * DIMC + hh * HD;

  // ---- stage Q,K,V,Vs,Vh tiles into LDS ----------------------------------
#if HAVE_TDM
  if (tid < 32) {                       // wave 0 drives the TDM
    tdm_load_tile64x64(q  + base, Ql);
    tdm_load_tile64x64(kk + base, Kl);
    tdm_load_tile64x64(v  + base, Vrow[0]);
    tdm_load_tile64x64(vs + base, Vrow[1]);
    tdm_load_tile64x64(vh + base, Vrow[2]);
    __builtin_amdgcn_s_wait_tensorcnt(3);   // Q and K landed
  }
#else
  for (int e = tid; e < 512; e += 128) {    // 64 rows x 8 chunks of 8 elems
    int m = e >> 3, c = (e & 7) * 8;
    const size_t g = base + (size_t)m * DIMC + c;
    *(v8bf*)&Ql[m * LPAD + c]      = *(const v8bf*)(q  + g);
    *(v8bf*)&Kl[m * LPAD + c]      = *(const v8bf*)(kk + g);
    *(v8bf*)&Vrow[0][m * LPAD + c] = *(const v8bf*)(v  + g);
    *(v8bf*)&Vrow[1][m * LPAD + c] = *(const v8bf*)(vs + g);
    *(v8bf*)&Vrow[2][m * LPAD + c] = *(const v8bf*)(vh + g);
  }
#endif
  __syncthreads();

  // ---- S = q @ k^T (from LDS) -------------------------------------------
  v8f acc[4] = {};
  const int mrow = wave * 16 + l15;
  for (int k0 = 0; k0 < HD; k0 += 32) {
    const bf16_t* pa = &Ql[mrow * LPAD + k0 + lhalf * 8];
    v16bf af = cat16(*(const v8bf*)pa, *(const v8bf*)(pa + 16));
#pragma unroll
    for (int nt = 0; nt < 4; ++nt) {
      const bf16_t* pb = &Kl[(nt * 16 + l15) * LPAD + k0 + lhalf * 8];
      v16bf bfv = cat16(*(const v8bf*)pb, *(const v8bf*)(pb + 16));
      acc[nt] = __builtin_amdgcn_wmma_f32_16x16x32_bf16(
          false, af, false, bfv, (short)0, acc[nt], false, false);
    }
  }

  // ---- + relative position bias + shifted-window mask --------------------
  const int wi  = w & 63;
  const int wwh = wi >> 3, www = wi & 7;
#pragma unroll
  for (int nt = 0; nt < 4; ++nt)
#pragma unroll
    for (int r = 0; r < 8; ++r) {
      int i = wave * 16 + lhalf * 8 + r;   // query token
      int j = nt * 16 + l15;               // key token
      int ridx = (((i >> 3) - (j >> 3)) + 7) * 15 + ((i & 7) - (j & 7) + 7);
      float f = acc[nt][r] + rpb[ridx * HEADS + hh];
      int li = region(wwh * 8 + (i >> 3)) * 3 + region(www * 8 + (i & 7));
      int lj = region(wwh * 8 + (j >> 3)) * 3 + region(www * 8 + (j & 7));
      if (li != lj) f -= 100.0f;
      S[i * SPAD + j] = f;
    }
  __syncthreads();

  // ---- softmax over rows (thread t owns row t, t < 64) -------------------
  if (tid < NWIN) {
    float mx = -3.4e38f;
    for (int j = 0; j < NWIN; ++j) mx = fmaxf(mx, S[tid * SPAD + j]);
    float sum = 0.0f;
    for (int j = 0; j < NWIN; ++j) {
      float e = __expf(S[tid * SPAD + j] - mx);
      S[tid * SPAD + j] = e;
      sum += e;
    }
    float inv = 1.0f / sum;
    for (int j = 0; j < NWIN; ++j)
      P[tid * LPAD + j] = (bf16_t)(S[tid * SPAD + j] * inv);
  }
#if HAVE_TDM
  if (tid < 32) __builtin_amdgcn_s_wait_tensorcnt(0);  // all V tiles landed
#endif
  __syncthreads();

  // ---- O = P @ v for each of v, vs, vh -----------------------------------
  bf16_t* optr[3] = {o, os, oh};
  for (int t = 0; t < 3; ++t) {
    // transpose Vrow[t] -> VT (LDS->LDS)
    for (int e = tid; e < NWIN * NWIN; e += 128) {
      int m = e >> 6, d = e & 63;
      VT[d * LPAD + m] = Vrow[t][m * LPAD + d];
    }
    __syncthreads();

    v8f oacc[4] = {};
    for (int k0 = 0; k0 < NWIN; k0 += 32) {
      const bf16_t* pa = &P[(wave * 16 + l15) * LPAD + k0 + lhalf * 8];
      v16bf af = cat16(*(const v8bf*)pa, *(const v8bf*)(pa + 16));
#pragma unroll
      for (int nt = 0; nt < 4; ++nt) {
        const bf16_t* pb = &VT[(nt * 16 + l15) * LPAD + k0 + lhalf * 8];
        v16bf bfv = cat16(*(const v8bf*)pb, *(const v8bf*)(pb + 16));
        oacc[nt] = __builtin_amdgcn_wmma_f32_16x16x32_bf16(
            false, af, false, bfv, (short)0, oacc[nt], false, false);
      }
    }
#pragma unroll
    for (int nt = 0; nt < 4; ++nt)
#pragma unroll
      for (int r = 0; r < 8; ++r) {
        int n = wave * 16 + lhalf * 8 + r;
        int d = nt * 16 + l15;
        optr[t][base + (size_t)n * DIMC + d] = (bf16_t)oacc[nt][r];
      }
    __syncthreads();   // VT reused next iteration
  }
}

// ---------------------------------------------------------------------------
// Host orchestration
// ---------------------------------------------------------------------------
static void launch_gemm(hipStream_t s, const bf16_t* A, const bf16_t* WT,
                        const float* bias, int M, int K, int Nout,
                        bf16_t* outB, float* outF, const float* resid,
                        int mode, int relu, float oscale) {
  dim3 g(Nout / 256, M / 64);
  k_gemm<<<g, 256, 0, s>>>(A, WT, bias, K, Nout, outB, outF, resid, mode, relu, oscale);
}

extern "C" void kernel_launch(void* const* d_in, const int* in_sizes, int n_in,
                              void* d_out, int out_size, void* d_ws, size_t ws_size,
                              hipStream_t stream) {
  (void)in_sizes; (void)n_in; (void)out_size; (void)ws_size;

  const float* xin   = (const float*)d_in[0];
  const float* sclin = (const float*)d_in[1];
  const float* shfin = (const float*)d_in[2];
  const float *W[8], *Bv[8];
  for (int i = 0; i < 8; ++i) { W[i] = (const float*)d_in[3 + 2*i]; Bv[i] = (const float*)d_in[4 + 2*i]; }
  const float* rpb = (const float*)d_in[19];
  const float *W1[3], *B1[3], *W2[3], *B2[3];
  for (int j = 0; j < 3; ++j) {
    W1[j] = (const float*)d_in[20 + 4*j]; B1[j] = (const float*)d_in[21 + 4*j];
    W2[j] = (const float*)d_in[22 + 4*j]; B2[j] = (const float*)d_in[23 + 4*j];
  }

  float* outX[3];
  outX[0] = (float*)d_out;
  outX[1] = outX[0] + (size_t)MTOT * DIMC;
  outX[2] = outX[1] + (size_t)MTOT * DIMC;
  const float* residIn[3] = {xin, sclin, shfin};

  // ---- workspace carve-up (~592 MB) --------------------------------------
  char* ws = (char*)d_ws;
  size_t off = 0;
  auto carve = [&](size_t bytes) -> char* {
    char* p = ws + off; off += (bytes + 255) & ~(size_t)255; return p;
  };
  bf16_t* wt[8];
  for (int i = 0; i < 8; ++i) wt[i] = (bf16_t*)carve((size_t)DIMC * DIMC * 2);
  bf16_t *wt1[3], *wt2[3];
  for (int j = 0; j < 3; ++j) wt1[j] = (bf16_t*)carve((size_t)HID * DIMC * 2);
  for (int j = 0; j < 3; ++j) wt2[j] = (bf16_t*)carve((size_t)DIMC * HID * 2);

  const size_t TBYTES = (size_t)MTOT * DIMC * 2;          // 64 MB per bf16 tensor
  char* bufA = carve((size_t)MTOT * HID * 2);             // 256 MB
  char* bufB = carve(5 * TBYTES);                         // 320 MB

  // bufA phase 1: xw/sw/hw -> phase 2: o/os/oh -> phase 3: MLP hidden
  bf16_t* xw = (bf16_t*)(bufA);
  bf16_t* sw = (bf16_t*)(bufA + TBYTES);
  bf16_t* hw = (bf16_t*)(bufA + 2 * TBYTES);
  bf16_t* ob[3] = {xw, sw, hw};            // attention outputs reuse same slots
  bf16_t* hid = (bf16_t*)bufA;             // MLP hidden reuses all of bufA
  // bufB phase 1: q,k,v,vs,vh -> phase 2: x1b,s1b,h1b
  bf16_t* qb  = (bf16_t*)(bufB);
  bf16_t* kb  = (bf16_t*)(bufB + TBYTES);
  bf16_t* vb  = (bf16_t*)(bufB + 2 * TBYTES);
  bf16_t* vsb = (bf16_t*)(bufB + 3 * TBYTES);
  bf16_t* vhb = (bf16_t*)(bufB + 4 * TBYTES);
  bf16_t* x1b[3] = {qb, kb, vb};

  // ---- 1. weight convert + transpose -------------------------------------
  for (int i = 0; i < 8; ++i)
    k_wt<<<(DIMC * DIMC + 255) / 256, 256, 0, stream>>>(W[i], wt[i], DIMC, DIMC);
  for (int j = 0; j < 3; ++j)
    k_wt<<<(DIMC * HID + 255) / 256, 256, 0, stream>>>(W1[j], wt1[j], DIMC, HID);
  for (int j = 0; j < 3; ++j)
    k_wt<<<(HID * DIMC + 255) / 256, 256, 0, stream>>>(W2[j], wt2[j], HID, DIMC);

  // ---- 2. shift + window partition ---------------------------------------
  const int PELEMS = MTOT * DIMC;
  k_part<<<PELEMS / 256, 256, 0, stream>>>(xin,   xw);
  k_part<<<PELEMS / 256, 256, 0, stream>>>(sclin, sw);
  k_part<<<PELEMS / 256, 256, 0, stream>>>(shfin, hw);

  // ---- 3. qkv projections (SC folded into q) -----------------------------
  launch_gemm(stream, xw, wt[0], Bv[0], MTOT, DIMC, DIMC, qb,  nullptr, nullptr, 0, 0, SCQ);
  launch_gemm(stream, xw, wt[1], Bv[1], MTOT, DIMC, DIMC, kb,  nullptr, nullptr, 0, 0, 1.0f);
  launch_gemm(stream, xw, wt[2], Bv[2], MTOT, DIMC, DIMC, vb,  nullptr, nullptr, 0, 0, 1.0f);
  launch_gemm(stream, sw, wt[3], Bv[3], MTOT, DIMC, DIMC, vsb, nullptr, nullptr, 0, 0, 1.0f);
  launch_gemm(stream, hw, wt[4], Bv[4], MTOT, DIMC, DIMC, vhb, nullptr, nullptr, 0, 0, 1.0f);

  // ---- 4. windowed attention (1024 windows x 8 heads) --------------------
  k_attn<<<BATCH * NWPB * HEADS, 128, 0, stream>>>(qb, kb, vb, vsb, vhb, rpb,
                                                   ob[0], ob[1], ob[2]);

  // ---- 5. output projection + window-reverse + residual ------------------
  for (int j = 0; j < 3; ++j)
    launch_gemm(stream, ob[j], wt[5 + j], Bv[5 + j], MTOT, DIMC, DIMC,
                x1b[j], outX[j], residIn[j], /*mode=*/1, 0, 1.0f);

  // ---- 6. MLPs with residual ---------------------------------------------
  for (int j = 0; j < 3; ++j) {
    launch_gemm(stream, x1b[j], wt1[j], B1[j], MTOT, DIMC, HID,
                hid, nullptr, nullptr, 0, /*relu=*/1, 1.0f);
    launch_gemm(stream, hid, wt2[j], B2[j], MTOT, HID, DIMC,
                nullptr, outX[j], outX[j], 0, 0, 1.0f);
  }
}
